// recurrent_encoder_68401649156712
// MI455X (gfx1250) — matrix operand order repaired and checked by reading, not compiled, not run on hardware
//
#include <hip/hip_runtime.h>
#include <hip/hip_bf16.h>

typedef __bf16 bf16_t;
typedef __attribute__((ext_vector_type(16))) __bf16 v16bf;
typedef __attribute__((ext_vector_type(8)))  __bf16 v8bf;
typedef __attribute__((ext_vector_type(8)))  float  v8f;

#define BB   64
#define TT   1024
#define DD   128
#define HH   256
#define G4H  1024
#define D2   512
#define HPAD 272   // padded LDS row stride (bf16 elems), keeps 16B alignment

#define LOG2E 1.4426950408889634f

__device__ __forceinline__ v16bf cat8(v8bf a, v8bf b) {
  v16bf r;
#pragma unroll
  for (int i = 0; i < 8; ++i) { r[i] = a[i]; r[8 + i] = b[i]; }
  return r;
}
// Branchless hardware-transcendental activations (v_exp_f32 + v_rcp_f32).
__device__ __forceinline__ float fast_sig(float x) {
  return __builtin_amdgcn_rcpf(1.f + __builtin_amdgcn_exp2f(-LOG2E * x));
}
__device__ __forceinline__ float fast_tanh(float x) {
  return 1.f - 2.f * __builtin_amdgcn_rcpf(1.f + __builtin_amdgcn_exp2f(2.f * LOG2E * x));
}

// ---------------------------------------------------------------- fp32 -> bf16
__global__ void cvt_bf16(const float* __restrict__ in, bf16_t* __restrict__ out, int n) {
  int i = blockIdx.x * blockDim.x + threadIdx.x;
  if (i < n) out[i] = (bf16_t)in[i];
}

// -------------------------------------------------- xg = X @ W^T + bih + bhh
// X: [M=B*T, K] bf16 row-major (row = b*T + t), W: [4H, K] bf16 row-major.
// Output XG stored [T][B][4H] fp32 so the scan reads one contiguous slab per t.
template <int K>
__global__ __launch_bounds__(256)
void xproj_wmma(const bf16_t* __restrict__ X, const bf16_t* __restrict__ W,
                const float* __restrict__ bih, const float* __restrict__ bhh,
                float* __restrict__ XG) {
  const int lane = threadIdx.x & 31;
  const int wave = threadIdx.x >> 5;
  const int mt   = blockIdx.y * 8 + wave;   // M-tile (16 rows)
  const int nt   = blockIdx.x;              // N-tile (0..63)
  const int nlo  = lane & 15;
  const int hi   = lane >> 4;
  const int rA   = mt * 16 + nlo;           // A row for this lane
  const int nB   = nt * 16 + nlo;           // B column for this lane
  const int koffA = hi * 16;
  const int koffB = hi * 16;

  v8f acc;
  const float bias = bih[nB] + bhh[nB];
#pragma unroll
  for (int v = 0; v < 8; ++v) acc[v] = bias;

#pragma unroll 4
  for (int k0 = 0; k0 < K; k0 += 32) {
    const bf16_t* ap = X + (size_t)rA * K + k0 + koffA;
    v16bf afrag = cat8(*(const v8bf*)ap, *(const v8bf*)(ap + 8));
    v16bf bfrag = *(const v16bf*)(W + (size_t)nB * K + k0 + koffB);
    acc = __builtin_amdgcn_wmma_f32_16x16x32_bf16(false, afrag, false, bfrag,
                                                  (short)0, acc, false, false);
  }
  const int rbase = mt * 16 + hi * 8;
#pragma unroll
  for (int v = 0; v < 8; ++v) {
    int r = rbase + v;
    int b = r >> 10;          // / T
    int t = r & (TT - 1);     // % T
    XG[((size_t)t * BB + b) * G4H + nt * 16 + nlo] = acc[v];
  }
}

// ------------------------------------------------------------- LSTM recurrence
// One workgroup (16 waves) runs the whole scan for one (layer, direction).
// Wave w owns hidden columns [16w,16w+16) across all 4 gates -> 16 WMMA tiles,
// processed as two M-halves of 2 tiles each.
// W_hh B-fragments are re-streamed from L2 every step:
//  - asm memory clobber stops LICM from hoisting them out of the time loop;
//  - the k-loop is kept ROLLED (unroll 1) so only one k-step's fragments are
//    live (~170 VGPRs total) -> no scratch spills on the critical scan path.
__global__ __launch_bounds__(512)
void lstm_rec(const float* __restrict__ XG,   // [T][B][4H] fp32 (biases folded)
              const bf16_t* __restrict__ WHH, // [4H][H] bf16 row-major
              bf16_t* __restrict__ Ybf,       // [B*T][2H] bf16 (or null)
              float*  __restrict__ Yf,        // [B*T][2H] fp32 (or null)
              float*  __restrict__ Hout,      // [B][H] final hidden
              int dircol, int reverse) {
  __shared__ bf16_t hsh[BB * HPAD];           // h(t-1), bf16, 34 KB
  const int lane = threadIdx.x & 31;
  const int wave = threadIdx.x >> 5;          // 0..15
  const int j0   = wave * 16;                 // hidden column base
  const int nlo  = lane & 15;
  const int hi   = lane >> 4;
  const int koffA = hi * 16;
  const int koffB = hi * 16;
  const int j = j0 + nlo;

  for (int i = threadIdx.x; i < BB * HPAD; i += 512) hsh[i] = (bf16_t)0.f;
  float c[4][8];
#pragma unroll
  for (int m = 0; m < 4; ++m)
#pragma unroll
    for (int v = 0; v < 8; ++v) c[m][v] = 0.f;
  __syncthreads();

  for (int tt = 0; tt < TT; ++tt) {
    asm volatile("" ::: "memory");  // keep W_hh fragment loads inside the loop
    const int t = reverse ? (TT - 1 - tt) : tt;
    const float* xgt = XG + (size_t)t * BB * G4H;
    if (tt + 1 < TT) {  // prefetch next timestep's gate slab (L2 -> near)
      const int tn = reverse ? (TT - 2 - tt) : (tt + 1);
      const float* xgn = XG + (size_t)tn * BB * G4H;
#pragma unroll
      for (int p = 0; p < 4; ++p)
        __builtin_prefetch(xgn + (((size_t)p * 512 + threadIdx.x) << 5), 0, 0);
    }

    for (int mh = 0; mh < 2; ++mh) {          // two halves of the batch dim
      v8f acc[4][2];                          // [gate][m-in-half]
#pragma unroll
      for (int g = 0; g < 4; ++g) {
        const int col = g * HH + j;
#pragma unroll
        for (int mi = 0; mi < 2; ++mi) {
          const int rb = (mh * 2 + mi) * 16 + hi * 8;
#pragma unroll
          for (int v = 0; v < 8; ++v)
            acc[g][mi][v] = xgt[(size_t)(rb + v) * G4H + col];
        }
      }
#pragma unroll 1
      for (int ki = 0; ki < 8; ++ki) {        // K = 256 = 8 x 32 (kept rolled!)
        const int k0 = ki * 32;
        v16bf a[2];
#pragma unroll
        for (int mi = 0; mi < 2; ++mi) {
          const bf16_t* ap = hsh + ((mh * 2 + mi) * 16 + nlo) * HPAD + k0 + koffA;
          a[mi] = cat8(*(const v8bf*)ap, *(const v8bf*)(ap + 8));
        }
#pragma unroll
        for (int g = 0; g < 4; ++g) {
          v16bf bfrag = *(const v16bf*)(WHH + (size_t)(g * HH + j) * HH + k0 + koffB);
#pragma unroll
          for (int mi = 0; mi < 2; ++mi)
            acc[g][mi] = __builtin_amdgcn_wmma_f32_16x16x32_bf16(
                false, a[mi], false, bfrag, (short)0, acc[g][mi], false, false);
        }
      }
      __syncthreads();                        // all reads of this half's h rows done
#pragma unroll
      for (int mi = 0; mi < 2; ++mi) {
        const int m  = mh * 2 + mi;
        const int rb = m * 16 + hi * 8;
#pragma unroll
        for (int v = 0; v < 8; ++v) {
          float i_ = fast_sig(acc[0][mi][v]);
          float f_ = fast_sig(acc[1][mi][v]);
          float g_ = fast_tanh(acc[2][mi][v]);
          float o_ = fast_sig(acc[3][mi][v]);
          float cv = f_ * c[m][v] + i_ * g_;
          c[m][v] = cv;
          float hv = o_ * fast_tanh(cv);
          const int b = rb + v;
          hsh[b * HPAD + j] = (bf16_t)hv;
          const size_t yrow = (size_t)b * TT + t;
          if (Ybf) Ybf[yrow * D2 + dircol + j] = (bf16_t)hv;
          if (Yf)  Yf [yrow * D2 + dircol + j] = hv;
          if (tt == TT - 1) Hout[b * HH + j] = hv;   // h_T only
        }
      }
      __syncthreads();                        // h(t) visible before next reads
    }
  }
}

// -------------------------------------------------- attention score + softmax
// One wave per output row: lanes split the 512-d dot (coalesced), shfl-reduce.
__global__ void scores_k(const float* __restrict__ Y, const float* __restrict__ ctx,
                         float* __restrict__ S) {
  const int lane = threadIdx.x & 31;
  const int wv   = threadIdx.x >> 5;          // 0..7
  const int t    = blockIdx.x;
  for (int b = wv; b < BB; b += 8) {
    const float* row = Y + ((size_t)b * TT + t) * D2;
    float s = 0.f;
    for (int d = lane; d < D2; d += 32) s += row[d] * ctx[d];
#pragma unroll
    for (int off = 16; off > 0; off >>= 1) s += __shfl_down(s, off, 32);
    if (lane == 0) S[t * BB + b] = s;
  }
}

__global__ void softmax_k(const float* __restrict__ S, float* __restrict__ alpha) {
  __shared__ float red[BB];
  const int t = blockIdx.x, b = threadIdx.x;
  float v = S[t * BB + b];
  red[b] = v; __syncthreads();
  for (int s = 32; s > 0; s >>= 1) { if (b < s) red[b] = fmaxf(red[b], red[b + s]); __syncthreads(); }
  float mx = red[0]; __syncthreads();
  float e = expf(v - mx);
  red[b] = e; __syncthreads();
  for (int s = 32; s > 0; s >>= 1) { if (b < s) red[b] += red[b + s]; __syncthreads(); }
  alpha[t * BB + b] = e / red[0];
}

// repres[b,d] = sum_j alpha_flat[b*T+j] * iput_flat[b*T+j, d]  (raw-reshape view)
__global__ void repres_k(const float* __restrict__ Y, const float* __restrict__ alpha,
                         float* __restrict__ R) {
  const int b = blockIdx.x, d = threadIdx.x;  // 64 blocks x 512 threads
  float s = 0.f;
  for (int jj = 0; jj < TT; ++jj) {
    int k  = b * TT + jj;
    int tp = k >> 6;        // flat row -> t'
    int bp = k & 63;        //          -> b'
    s += alpha[k] * Y[((size_t)bp * TT + tp) * D2 + d];
  }
  R[b * D2 + d] = s;
}

// ------------------------------------------------------------------ dispatch
extern "C" void kernel_launch(void* const* d_in, const int* in_sizes, int n_in,
                              void* d_out, int out_size, void* d_ws, size_t ws_size,
                              hipStream_t stream) {
  (void)in_sizes; (void)n_in; (void)out_size; (void)ws_size;
  const float* seq = (const float*)d_in[0];
  const float* wih[4] = {(const float*)d_in[1], (const float*)d_in[5],
                         (const float*)d_in[9], (const float*)d_in[13]};
  const float* whh[4] = {(const float*)d_in[2], (const float*)d_in[6],
                         (const float*)d_in[10], (const float*)d_in[14]};
  const float* bih[4] = {(const float*)d_in[3], (const float*)d_in[7],
                         (const float*)d_in[11], (const float*)d_in[15]};
  const float* bhh[4] = {(const float*)d_in[4], (const float*)d_in[8],
                         (const float*)d_in[12], (const float*)d_in[16]};
  const float* ctx = (const float*)d_in[17];

  char* w = (char*)d_ws;
  float*  xg   = (float*)w;  w += (size_t)TT * BB * G4H * 4;   // 256 MB
  bf16_t* ybf  = (bf16_t*)w; w += (size_t)BB * TT * D2 * 2;    //  64 MB
  float*  y1   = (float*)w;  w += (size_t)BB * TT * D2 * 4;    // 128 MB
  bf16_t* seqb = (bf16_t*)w; w += (size_t)BB * TT * DD * 2;    //  16 MB
  bf16_t* wihb[4]; bf16_t* whhb[4];
  const int wihN[4] = {G4H * DD, G4H * DD, G4H * D2, G4H * D2};
  for (int i = 0; i < 4; ++i) { wihb[i] = (bf16_t*)w; w += (size_t)wihN[i] * 2; }
  for (int i = 0; i < 4; ++i) { whhb[i] = (bf16_t*)w; w += (size_t)G4H * HH * 2; }
  float* sc = (float*)w;     w += (size_t)TT * BB * 4;

  float* out_repres = (float*)d_out;                 // [64, 1, 512]
  float* out_hid    = out_repres + BB * D2;          // [4][64][256] raw
  float* out_alpha  = out_hid + 4 * BB * HH;         // [T][B]

  // 0) bf16 conversions
  {
    int n = BB * TT * DD;
    cvt_bf16<<<(n + 255) / 256, 256, 0, stream>>>(seq, seqb, n);
    for (int i = 0; i < 4; ++i) {
      cvt_bf16<<<(wihN[i] + 255) / 256, 256, 0, stream>>>(wih[i], wihb[i], wihN[i]);
      int m = G4H * HH;
      cvt_bf16<<<(m + 255) / 256, 256, 0, stream>>>(whh[i], whhb[i], m);
    }
  }

  dim3 gGemm(G4H / 16, (BB * TT) / 128);
  // 1) layer 0, both directions (xg buffer reused)
  for (int dir = 0; dir < 2; ++dir) {
    xproj_wmma<DD><<<gGemm, 256, 0, stream>>>(seqb, wihb[dir], bih[dir], bhh[dir], xg);
    lstm_rec<<<1, 512, 0, stream>>>(xg, whhb[dir], ybf, nullptr,
                                    out_hid + dir * BB * HH, dir * HH, dir);
  }
  // 2) layer 1, both directions (input = bf16 concat of layer-0 outputs)
  for (int dir = 0; dir < 2; ++dir) {
    xproj_wmma<D2><<<gGemm, 256, 0, stream>>>(ybf, wihb[2 + dir], bih[2 + dir], bhh[2 + dir], xg);
    lstm_rec<<<1, 512, 0, stream>>>(xg, whhb[2 + dir], nullptr, y1,
                                    out_hid + (2 + dir) * BB * HH, dir * HH, dir);
  }
  // 3) attention tail
  scores_k <<<TT, 256, 0, stream>>>(y1, ctx, sc);
  softmax_k<<<TT, BB, 0, stream>>>(sc, out_alpha);
  repres_k <<<BB, D2, 0, stream>>>(y1, out_alpha, out_repres);
}
